// BlackBox_14877766713677
// MI455X (gfx1250) — matrix operand order repaired and checked by reading, not compiled, not run on hardware
//
#include <hip/hip_runtime.h>
#include <hip/hip_bf16.h>
#include <math.h>

typedef __attribute__((ext_vector_type(16))) _Float16 v16h;
typedef __attribute__((ext_vector_type(8)))  _Float16 v8h;
typedef __attribute__((ext_vector_type(8)))  float    v8f;

#define D      256
#define SP     264   // padded LDS row pitch in halves: 528 B = 33*16 B (16B-aligned rows,
                     // rows land 4 banks apart -> conflict-free b128 fragment loads)
#define BQ     8
#define NQ     512
#define NSTEPS 12
#define VOCABN 32000

// ---- WMMA fragment builders (contiguous-k lane layouts used by this file) ----------
// A (16x32 f16): lane L: M = L%16; elems 0..15 -> K = base+hs*16+{0..15}
static __device__ __forceinline__ v16h make_a(const _Float16* p) {
  v8h lo = *(const v8h*)p;
  v8h hi = *(const v8h*)(p + 8);
  v16h a;
#pragma unroll
  for (int e = 0; e < 8; ++e) { a[e] = lo[e]; a[8 + e] = hi[e]; }
  return a;
}
// B (32x16 f16): lane L: N = L%16; elems 0..15 -> K = base+hs*16+{0..15}
static __device__ __forceinline__ v16h make_b(const _Float16* p) {
  v8h lo = *(const v8h*)p;
  v8h hi = *(const v8h*)(p + 8);
  v16h b;
#pragma unroll
  for (int e = 0; e < 8; ++e) { b[e] = lo[e]; b[8 + e] = hi[e]; }
  return b;
}

// Branchless exact-style GELU: 0.5*z*(1+erf(z/sqrt2)), erf via A&S 7.1.26
// (|err| < 1.5e-7 — below the f16 state-rounding noise floor; no divergence,
//  uses v_rcp_f32 + v_exp_f32 transcendentals, ~13 straight-line VALU ops)
static __device__ __forceinline__ float gelu_f(float z) {
  const float x  = z * 0.70710678118654752f;
  const float ax = fabsf(x);
  const float t  = __builtin_amdgcn_rcpf(fmaf(0.3275911f, ax, 1.0f));
  float p = fmaf(t, 1.061405429f, -1.453152027f);
  p = fmaf(t, p, 1.421413741f);
  p = fmaf(t, p, -0.284496736f);
  p = fmaf(t, p, 0.254829592f);
  const float e  = __expf(-ax * ax);
  const float er = fmaf(-p * t, e, 1.0f);           // erf(|x|)
  const float erfv = copysignf(er, x);
  return 0.5f * z * (1.0f + erfv);
}

// ---- Kernel 1: sequential recurrent scan, single workgroup ---------------
__global__ __launch_bounds__(512) void scan_kernel(
    const int*   __restrict__ ids,        // [BQ, NQ]
    const float* __restrict__ embed,      // [VOCABN, D]
    const float* __restrict__ W,          // [D, D]
    const float* __restrict__ pos_scale,  // [NSTEPS]
    _Float16*    __restrict__ states_out) // [BQ*NQ, D] f16
{
  __shared__ __align__(16) _Float16 sbuf[2][16 * SP]; // ping-pong state (M padded to 16)
  __shared__ float s_ps[NSTEPS];

  const int tid  = threadIdx.x;
  const int wave = tid >> 5;
  const int lane = tid & 31;
  const int hs   = (lane >> 4) & 1;
  const int nl   = lane & 15;
  const int n0   = wave * 16;     // 16 waves cover N = 0..255
  const int col  = n0 + nl;

  for (int i = tid; i < 16 * SP; i += 512) {
    sbuf[0][i] = (_Float16)0.f;
    sbuf[1][i] = (_Float16)0.f;
  }
  if (tid < NSTEPS) s_ps[tid] = pos_scale[tid];

  // Keep this wave's W tile in registers: B[k][n] = W[n][k] (out = state @ W^T)
  v16h breg[8];
  {
    const float* wrow = W + (size_t)col * D;
#pragma unroll
    for (int kb = 0; kb < 8; ++kb) {
      const int kbase = kb * 32 + hs * 16;
#pragma unroll
      for (int e = 0; e < 16; ++e)
        breg[kb][e] = (_Float16)wrow[kbase + e];
    }
  }
  __syncthreads();

  const int erow = tid >> 6;        // 0..7  (batch row)
  const int ecol = (tid & 63) * 4;  // 0..252

  // Register-carried per-lane state values at (m = r+hs*8, col):
  //   carry[r] = state value entering the current inner step (f32, like reference)
  //   prevv[r] = prev_state (final state of previous position)
  float prevv[8];
#pragma unroll
  for (int r = 0; r < 8; ++r) prevv[r] = 0.0f;

#pragma unroll 1
  for (int pos = 0; pos < NQ; ++pos) {
    // Embedding gather into sbuf[0], rows 0..7 (rows 8..15 stay zero by induction)
    {
      const int id = ids[erow * NQ + pos];
      const float4 v = *(const float4*)(embed + (size_t)id * D + ecol);
      _Float16* dst = &sbuf[0][erow * SP + ecol];
      dst[0] = (_Float16)v.x; dst[1] = (_Float16)v.y;
      dst[2] = (_Float16)v.z; dst[3] = (_Float16)v.w;
    }
    __syncthreads();

    float carry[8];
#pragma unroll
    for (int r = 0; r < 8; ++r)
      carry[r] = (float)sbuf[0][(r + hs * 8) * SP + col];

    const float ctx = (pos > 0) ? (1.0f / (float)(pos + 1)) : 0.0f;
    int p = 0;
#pragma unroll 1
    for (int s = 0; s < NSTEPS; ++s) {
      const _Float16* cur = &sbuf[p][0];
      _Float16*       nxt = &sbuf[p ^ 1][0];

      // Two accumulators: halves the WMMA C->C dependent chain (latency-bound loop)
      v8f acc0 = {}, acc1 = {};
#pragma unroll
      for (int kb = 0; kb < 8; kb += 2) {
        v16h a0 = make_a(cur + nl * SP + kb * 32 + hs * 16);
        v16h a1 = make_a(cur + nl * SP + (kb + 1) * 32 + hs * 16);
        acc0 = __builtin_amdgcn_wmma_f32_16x16x32_f16(
            false, a0, false, breg[kb], (short)0, acc0, false, false);
        acc1 = __builtin_amdgcn_wmma_f32_16x16x32_f16(
            false, a1, false, breg[kb + 1], (short)0, acc1, false, false);
      }

      const float ps   = s_ps[s];
      const bool  last = (s == NSTEPS - 1);
#pragma unroll
      for (int r = 0; r < 8; ++r) {
        const int m = r + hs * 8;            // C layout: M = r + hs*8, N = lane%16
        const float z   = (acc0[r] + acc1[r]) + ps * carry[r];
        const float res = gelu_f(z) + ctx * prevv[r];
        carry[r] = res;
        const _Float16 h = (_Float16)res;
        nxt[m * SP + col] = h;
        if (last) {
          prevv[r] = res;
          if (m < BQ)
            states_out[((size_t)m * NQ + pos) * D + col] = h;
        }
      }
      p ^= 1;
      __syncthreads();
    }
  }
}

// ---- Kernel 0: f32 -> f16 conversion of out_W ----------------------------
__global__ __launch_bounds__(256) void cvt_kernel(const float* __restrict__ src,
                                                  _Float16* __restrict__ dst,
                                                  int n4) {
  const int i = blockIdx.x * blockDim.x + threadIdx.x;
  if (i < n4) {
    const float4 v = *(const float4*)(src + (size_t)i * 4);
    _Float16* d = dst + (size_t)i * 4;
    d[0] = (_Float16)v.x; d[1] = (_Float16)v.y;
    d[2] = (_Float16)v.z; d[3] = (_Float16)v.w;
  }
}

// ---- Kernel 2: output projection GEMM (128M x 128N block, K=256 resident) -
__global__ __launch_bounds__(256) void gemm_kernel(
    const _Float16* __restrict__ A,    // [4096, 256] states f16
    const _Float16* __restrict__ Bw,   // [VOCABN, 256] out_W f16 (row n, K contiguous)
    const float*    __restrict__ bias, // [VOCABN]
    float*          __restrict__ out)  // [4096, VOCABN]
{
  extern __shared__ _Float16 smem[];   // 2 * 128 * SP halves = 132 KB
  _Float16* sA = smem;
  _Float16* sB = smem + 128 * SP;

  const int tid  = threadIdx.x;
  const int wave = tid >> 5, lane = tid & 31;
  const int hs   = (lane >> 4) & 1, nl = lane & 15;
  const int bm   = blockIdx.x, bn = blockIdx.y;

  {
    const uint4* gA = (const uint4*)(A  + (size_t)bm * 128 * D);
    const uint4* gB = (const uint4*)(Bw + (size_t)bn * 128 * D);
    uint4* lA = (uint4*)sA; uint4* lB = (uint4*)sB;  // padded pitch: 33 uint4 per row
#pragma unroll
    for (int t = 0; t < 16; ++t) {
      const int i = tid + 256 * t;           // i in [0,4096): row = i>>5, c16 = i&31
      lA[(i >> 5) * 33 + (i & 31)] = gA[i];
    }
#pragma unroll
    for (int t = 0; t < 16; ++t) {
      const int i = tid + 256 * t;
      lB[(i >> 5) * 33 + (i & 31)] = gB[i];
    }
  }
  __syncthreads();

  const int wm = wave & 3;   // 4-way M split  -> 32 rows
  const int wn = wave >> 2;  // 2-way N split  -> 64 cols

  v8f acc[2][4];
#pragma unroll
  for (int i = 0; i < 2; ++i)
#pragma unroll
    for (int j = 0; j < 4; ++j) acc[i][j] = (v8f){};

#pragma unroll
  for (int kb = 0; kb < 8; ++kb) {
    v16h af[2], bf[4];
#pragma unroll
    for (int i = 0; i < 2; ++i)
      af[i] = make_a(sA + (wm * 32 + i * 16 + nl) * SP + kb * 32 + hs * 16);
#pragma unroll
    for (int j = 0; j < 4; ++j)
      bf[j] = make_b(sB + (wn * 64 + j * 16 + nl) * SP + kb * 32 + hs * 16);
#pragma unroll
    for (int i = 0; i < 2; ++i)
#pragma unroll
      for (int j = 0; j < 4; ++j)
        acc[i][j] = __builtin_amdgcn_wmma_f32_16x16x32_f16(
            false, af[i], false, bf[j], (short)0, acc[i][j], false, false);
  }

  const int m0  = bm * 128 + wm * 32;
  const int n0g = bn * 128 + wn * 64;
#pragma unroll
  for (int j = 0; j < 4; ++j) {
    const int   n  = n0g + j * 16 + nl;
    const float bv = bias[n];
#pragma unroll
    for (int i = 0; i < 2; ++i) {
#pragma unroll
      for (int r = 0; r < 8; ++r) {
        const int m = m0 + i * 16 + r + hs * 8;
        out[(size_t)m * VOCABN + n] = acc[i][j][r] + bv;
      }
    }
  }
}

// ---- Host launcher --------------------------------------------------------
extern "C" void kernel_launch(void* const* d_in, const int* in_sizes, int n_in,
                              void* d_out, int out_size, void* d_ws, size_t ws_size,
                              hipStream_t stream) {
  const int*   ids       = (const int*)d_in[0];
  const float* embed     = (const float*)d_in[1];
  const float* W         = (const float*)d_in[2];
  const float* pos_scale = (const float*)d_in[3];
  const float* out_W     = (const float*)d_in[4];
  const float* out_b     = (const float*)d_in[5];
  float*       out       = (float*)d_out;

  _Float16* outW_h   = (_Float16*)d_ws;                                    // 16,384,000 B
  _Float16* states_h = (_Float16*)((char*)d_ws +
                                   (size_t)VOCABN * D * sizeof(_Float16)); // +2,097,152 B

  // 1) out_W f32 -> f16
  {
    const int n4 = VOCABN * D / 4;
    cvt_kernel<<<(n4 + 255) / 256, 256, 0, stream>>>(out_W, outW_h, n4);
  }
  // 2) recurrent scan (one WGP, 16 waves, W resident in registers)
  scan_kernel<<<1, 512, 0, stream>>>(ids, embed, W, pos_scale, states_h);
  // 3) output projection (HBM-roofline bound; 132KB dynamic LDS)
  {
    const int smem_bytes = 2 * 128 * SP * (int)sizeof(_Float16); // 135168
    hipFuncSetAttribute((const void*)gemm_kernel,
                        hipFuncAttributeMaxDynamicSharedMemorySize, smem_bytes);
    dim3 grid((BQ * NQ) / 128, VOCABN / 128);
    gemm_kernel<<<grid, 256, smem_bytes, stream>>>(states_h, outW_h, out_b, out);
  }
}
